// RSNN_36223754174476
// MI455X (gfx1250) — matrix operand order repaired and checked by reading, not compiled, not run on hardware
//
#include <hip/hip_runtime.h>

// ---------------------------------------------------------------------------
// RSNN forward for MI455X (gfx1250, wave32, WMMA), software-pipelined.
//
// 16 workgroups x 16 batch rows each run the full 512x4-step recurrence
// privately (batch rows are independent). 8 waves/WG; wave w owns hidden
// neuron tiles 4w..4w+3 (waves 0-3 static LIF, 4-7 adaptive ALIF; the split
// is wave-uniform and scalarized via readfirstlane) and output tile w.
// Neuron state (v_s/v_a/b_a/v_out) is register-resident in WMMA C-layout;
// spikes + converted input live in LDS as the f16 A-operand.
//
// GEMM inner loops are ping-pong pipelined: slice kt+1's A (ds_load_b128 x2)
// and B (global_load_b128 x2 per tile) are issued before slice kt's WMMAs,
// so loads overlap the 4 independent WMMA chains.
//
// Weights are pre-packed per launch into a contiguous-k B-operand lane layout (k = 16h + s, as in A):
// 32x16 (KxN) f16 tile -> lane n(0..15) holds col n, K 0..15 in V0..V7
// (pairs 2j,2j+1); lanes 16..31 hold K 16..31.
// ---------------------------------------------------------------------------

typedef __attribute__((ext_vector_type(16))) _Float16 v16h;
typedef __attribute__((ext_vector_type(8)))  float    v8f;
typedef __attribute__((ext_vector_type(4)))  float    f32x4;
typedef __attribute__((ext_vector_type(4)))  _Float16 h4;

#define T_STEPS 512
#define BSZ     256
#define NIN     256
#define NS      256
#define HID     512
#define NOUT    128
#define SIMT    4

#define BETA_D  0.9f
#define RHO_D   0.97f
#define THRESH  1.0f
#define BADAPT  1.8f
#define KAPPA   0.9f

#define MROWS    16
#define KTOT     (NIN + HID)     // 768
#define XSTRIDE  (KTOT + 8)      // 776 halves: 16B-aligned rows, bank-spread
#define NKT_HID  (KTOT / 32)     // 24
#define NKT_OUT  (HID / 32)      // 16
#define NT_HID   (HID / 16)      // 32
#define NT_OUT   (NOUT / 16)     // 8
#define TILE_H   (32 * 16)       // halves per 32x16 B tile = 512

#define WH_HALVES (NT_HID * NKT_HID * TILE_H)   // 768 KB packed hidden W
#define WO_HALVES (NT_OUT * NKT_OUT * TILE_H)   // 128 KB packed output W

static __device__ __forceinline__ v8f vzero8() {
    v8f z;
#pragma unroll
    for (int i = 0; i < 8; ++i) z[i] = 0.0f;
    return z;
}

// A-tile: lane m(=lane&15) holds row m; 16B loads at col hi*16 and hi*16+8.
#define LOAD_A(dst, base, kt) do {                                   \
    const _Float16* _ap = (base) + (kt) * 32;                        \
    ((f32x4*)&(dst))[0] = *(const f32x4*)(_ap);                      \
    ((f32x4*)&(dst))[1] = *(const f32x4*)(_ap + 8);                  \
} while (0)

// B-tile: lane reads its 16 contiguous pre-packed halves (2 x b128).
#define LOAD_B(dst, ptr) do {                                        \
    ((f32x4*)&(dst))[0] = *(const f32x4*)(ptr);                      \
    ((f32x4*)&(dst))[1] = *(const f32x4*)((ptr) + 8);                \
} while (0)

#define WMMA(acc, a, b)                                              \
    (acc) = __builtin_amdgcn_wmma_f32_16x16x32_f16(                  \
        false, (a), false, (b), (short)0, (acc), false, false)

// ---- weight pack kernels: gather f32 weights into B-operand f16 layout ----

__global__ void rsnn_pack_hidden(const float* __restrict__ Ws,
                                 const float* __restrict__ Wa,
                                 _Float16* __restrict__ dst) {
    int idx = blockIdx.x * blockDim.x + threadIdx.x;   // one thread per dword
    if (idx >= WH_HALVES / 2) return;
    int j    = idx & 7;
    int lane = (idx >> 3) & 31;
    int blk  = idx >> 8;            // ntile*24 + ktile
    int kt   = blk % NKT_HID;
    int nt   = blk / NKT_HID;
    int n    = nt * 16 + (lane & 15);
    int k    = kt * 32 + ((lane >> 4) << 4) + 2 * j;
    const float* W = (n < NS) ? Ws : Wa;
    int nn = (n < NS) ? n : (n - NS);
    dst[2 * idx]     = (_Float16)W[nn * KTOT + k];
    dst[2 * idx + 1] = (_Float16)W[nn * KTOT + k + 1];
}

__global__ void rsnn_pack_out(const float* __restrict__ Wo,
                              _Float16* __restrict__ dst) {
    int idx = blockIdx.x * blockDim.x + threadIdx.x;
    if (idx >= WO_HALVES / 2) return;
    int j    = idx & 7;
    int lane = (idx >> 3) & 31;
    int blk  = idx >> 8;            // ntile*16 + ktile
    int kt   = blk % NKT_OUT;
    int nt   = blk / NKT_OUT;
    int n    = nt * 16 + (lane & 15);
    int k    = kt * 32 + ((lane >> 4) << 4) + 2 * j;
    dst[2 * idx]     = (_Float16)Wo[n * HID + k];
    dst[2 * idx + 1] = (_Float16)Wo[n * HID + k + 1];
}

// ---------------------------- main RSNN kernel ----------------------------

__launch_bounds__(256, 1)
__global__ void rsnn_main(const float* __restrict__ inp,
                          const float* __restrict__ bs,
                          const float* __restrict__ ba,
                          const float* __restrict__ bo,
                          const _Float16* __restrict__ Wh,
                          const _Float16* __restrict__ Wop,
                          float* __restrict__ out) {
    __shared__ _Float16 xact[MROWS * XSTRIDE];   // [x_t(f16) | spikes(f16)]

    const int b0   = blockIdx.x * MROWS;
    const int tid  = threadIdx.x;
    // wave index is uniform across the wave: pin it into an SGPR so the
    // static/adaptive split compiles to a scalar branch (no exec masking).
    const int wv   = __builtin_amdgcn_readfirstlane(tid >> 5);   // 0..7
    const int lane = tid & 31;
    const int lrow = lane & 15;                  // N col (C-layout) / A row
    const int lhi  = lane >> 4;
    const bool adaptive = (wv >= 4);

    for (int i = tid; i < MROWS * XSTRIDE; i += 256) xact[i] = (_Float16)0.0f;

    // per-lane biases for 4 owned hidden tiles + 1 output tile
    float biash[4];
#pragma unroll
    for (int i = 0; i < 4; ++i) {
        int nt = wv * 4 + i;
        biash[i] = adaptive ? ba[(nt - 16) * 16 + lrow] : bs[nt * 16 + lrow];
    }
    const float biaso = bo[wv * 16 + lrow];

    // packed-weight base pointers (SGPR base + lane offset)
    const _Float16* bbh[4];
#pragma unroll
    for (int i = 0; i < 4; ++i)
        bbh[i] = Wh + (size_t)(wv * 4 + i) * NKT_HID * TILE_H + lane * 16;
    const _Float16* bbo = Wop + (size_t)wv * NKT_OUT * TILE_H + lane * 16;

    // register-resident state (C-layout: elem (M = r + 8*lhi, N = lrow))
    v8f vmem[4], bvar[4];
#pragma unroll
    for (int i = 0; i < 4; ++i) { vmem[i] = vzero8(); bvar[i] = vzero8(); }

    __syncthreads();

    const _Float16* arow   = xact + lrow * XSTRIDE + lhi * 16; // A-tile base
    const _Float16* srow   = arow + NIN;                       // spike A base
    _Float16* const spbase = xact + NIN;                       // spike region

    for (int t = 0; t < T_STEPS; ++t) {
        // ---- stage x_t into xact cols [0, NIN) as f16 (16 vals/thread) ----
        {
            const int row = tid >> 4;
            const int c0  = (tid & 15) * 16;
            const f32x4* src =
                (const f32x4*)(inp + ((size_t)t * BSZ + (b0 + row)) * NIN + c0);
            h4* dst = (h4*)(xact + row * XSTRIDE + c0);
#pragma unroll
            for (int c = 0; c < 4; ++c) {
                f32x4 v = src[c];
                h4 h;
                h[0] = (_Float16)v[0]; h[1] = (_Float16)v[1];
                h[2] = (_Float16)v[2]; h[3] = (_Float16)v[3];
                dst[c] = h;
            }
        }
        v8f vout = vzero8();                 // output integrator resets per t
        __syncthreads();

#pragma unroll 1
        for (int st = 0; st < SIMT; ++st) {
            // ------- hidden GEMM [16,768]x[768,4x16], ping-pong piped ------
            v8f acc[4];
#pragma unroll
            for (int i = 0; i < 4; ++i) acc[i] = vzero8();

            v16h ah[2], bh[2][4];
            LOAD_A(ah[0], arow, 0);
#pragma unroll
            for (int i = 0; i < 4; ++i) LOAD_B(bh[0][i], bbh[i]);

#pragma unroll 1
            for (int kt = 0; kt < NKT_HID; kt += 2) {
                LOAD_A(ah[1], arow, kt + 1);
#pragma unroll
                for (int i = 0; i < 4; ++i)
                    LOAD_B(bh[1][i], bbh[i] + (size_t)(kt + 1) * TILE_H);
#pragma unroll
                for (int i = 0; i < 4; ++i) WMMA(acc[i], ah[0], bh[0][i]);
                if (kt + 2 < NKT_HID) {
                    LOAD_A(ah[0], arow, kt + 2);
#pragma unroll
                    for (int i = 0; i < 4; ++i)
                        LOAD_B(bh[0][i], bbh[i] + (size_t)(kt + 2) * TILE_H);
                }
#pragma unroll
                for (int i = 0; i < 4; ++i) WMMA(acc[i], ah[1], bh[1][i]);
            }

            __syncthreads();   // all A-reads of old spikes done

            // ------------- neuron dynamics + spike write (f16) -------------
            if (adaptive) {
#pragma unroll
                for (int i = 0; i < 4; ++i) {
                    _Float16* sp = spbase + (wv * 4 + i) * 16 + lrow;
#pragma unroll
                    for (int r = 0; r < 8; ++r) {
                        float v = BETA_D * vmem[i][r] + acc[i][r] + biash[i];
                        float eff = THRESH + BADAPT * bvar[i][r];
                        float s = (v - eff) > 0.0f ? 1.0f : 0.0f;
                        v -= eff * s;
                        bvar[i][r] = RHO_D * bvar[i][r] + s;
                        vmem[i][r] = v;
                        sp[(r + lhi * 8) * XSTRIDE] = (_Float16)s;
                    }
                }
            } else {
#pragma unroll
                for (int i = 0; i < 4; ++i) {
                    _Float16* sp = spbase + (wv * 4 + i) * 16 + lrow;
#pragma unroll
                    for (int r = 0; r < 8; ++r) {
                        float v = BETA_D * vmem[i][r] + acc[i][r] + biash[i];
                        float s = (v - THRESH) > 0.0f ? 1.0f : 0.0f;
                        v -= THRESH * s;
                        vmem[i][r] = v;
                        sp[(r + lhi * 8) * XSTRIDE] = (_Float16)s;
                    }
                }
            }

            __syncthreads();   // new spikes visible

            // ------ output GEMM [16,512]x[512,16], piped, 2 acc chains -----
            v8f oacc0 = vzero8(), oacc1 = vzero8();
            v16h ao[2], bo2[2];
            LOAD_A(ao[0], srow, 0);
            LOAD_B(bo2[0], bbo);
#pragma unroll 1
            for (int kt = 0; kt < NKT_OUT; kt += 2) {
                LOAD_A(ao[1], srow, kt + 1);
                LOAD_B(bo2[1], bbo + (size_t)(kt + 1) * TILE_H);
                WMMA(oacc0, ao[0], bo2[0]);
                if (kt + 2 < NKT_OUT) {
                    LOAD_A(ao[0], srow, kt + 2);
                    LOAD_B(bo2[0], bbo + (size_t)(kt + 2) * TILE_H);
                }
                WMMA(oacc1, ao[1], bo2[1]);
            }
#pragma unroll
            for (int r = 0; r < 8; ++r)
                vout[r] = KAPPA * vout[r] + (oacc0[r] + oacc1[r]) + biaso;
        }

        // ---- emit out[t, b0+m, wv*16 + lrow] ----
#pragma unroll
        for (int r = 0; r < 8; ++r) {
            const int m = r + lhi * 8;
            out[((size_t)t * BSZ + b0 + m) * NOUT + wv * 16 + lrow] = vout[r];
        }
    }
}

// ------------------------------- launcher ---------------------------------

extern "C" void kernel_launch(void* const* d_in, const int* in_sizes, int n_in,
                              void* d_out, int out_size, void* d_ws, size_t ws_size,
                              hipStream_t stream) {
    const float* inp = (const float*)d_in[0];
    const float* Ws  = (const float*)d_in[1];
    const float* bs  = (const float*)d_in[2];
    const float* Wa  = (const float*)d_in[3];
    const float* ba  = (const float*)d_in[4];
    const float* Wo  = (const float*)d_in[5];
    const float* bo  = (const float*)d_in[6];
    float* out = (float*)d_out;

    _Float16* wh = (_Float16*)d_ws;             // 768 KB packed hidden weights
    _Float16* wo = wh + WH_HALVES;              // 128 KB packed output weights

    rsnn_pack_hidden<<<(WH_HALVES / 2 + 255) / 256, 256, 0, stream>>>(Ws, Wa, wh);
    rsnn_pack_out<<<(WO_HALVES / 2 + 255) / 256, 256, 0, stream>>>(Wo, wo);
    rsnn_main<<<BSZ / MROWS, 256, 0, stream>>>(inp, bs, ba, bo, wh, wo, out);
}
